// cross_att_fusion_block_83434034692672
// MI455X (gfx1250) — matrix operand order repaired and checked by reading, not compiled, not run on hardware
//
#include <hip/hip_runtime.h>

// B=1, C=64, N=65536, K=16. Gather is L2-resident (feature = 32MB << 192MB L2);
// projections mapped to v_wmma_f32_16x16x32_f16 (M=16 neighbors/points, N=16-ch
// tiles, K=32-ch steps, f32 accumulate).

typedef _Float16 f16;
typedef __attribute__((ext_vector_type(16))) _Float16 v16h;
typedef __attribute__((ext_vector_type(8)))  _Float16 v8h;
typedef __attribute__((ext_vector_type(8)))  float    v8f;

static __device__ __forceinline__ v16h cat8(v8h lo, v8h hi) {
  return __builtin_shufflevector(lo, hi, 0,1,2,3,4,5,6,7,8,9,10,11,12,13,14,15);
}

// A fragment (16x32 f16, per CDNA5 ISA layout): lane L holds row m=L%16;
// halves 0..7 = cols cbase+off8..+7, halves 8..15 = cols cbase+16+off8..(+7),
// with off8 = 8*(L/16). Rows are contiguous 256B in featT -> two 16B loads.
static __device__ __forceinline__ v16h load_afrag(const f16* __restrict__ row,
                                                  int cbase, int off8) {
  const v8h* p0 = (const v8h*)(row + cbase + off8);
  const v8h* p1 = (const v8h*)(row + cbase + 16 + off8);
  return cat8(*p0, *p1);
}

// B fragment (32x16 f16): B[k][n] = W[dtile+n][cbase+k] (proj = x * W^T).
// Lane L: n=L%16; halves 0..7 -> k = 8*(L/16)+h, halves 8..15 -> k = 16+8*(L/16)+(h-8)
// (same k order as the A fragment).
static __device__ __forceinline__ v16h load_bfrag(const f16* __restrict__ w,
                                                  int dtile, int lane16,
                                                  int hi16, int cbase) {
  const f16* r = w + (dtile + lane16) * 64 + cbase + (hi16 >> 1);
  const v8h* p0 = (const v8h*)r;
  const v8h* p1 = (const v8h*)(r + 16);
  return cat8(*p0, *p1);
}

// ---------------- prep kernels ----------------

__global__ void prep_weights(const float* __restrict__ WQa, const float* __restrict__ WKa,
                             const float* __restrict__ WVa, const float* __restrict__ WQe,
                             const float* __restrict__ WKe, const float* __restrict__ WVe,
                             f16* __restrict__ w16) {
  int i = blockIdx.x * blockDim.x + threadIdx.x;
  if (i >= 6 * 4096) return;
  int mi = i >> 12, j = i & 4095;
  const float* src = (mi == 0) ? WQa : (mi == 1) ? WKa : (mi == 2) ? WVa
                   : (mi == 3) ? WQe : (mi == 4) ? WKe : WVe;
  w16[i] = (f16)src[j];
}

// feature [128][N] f32 (channel-major) -> featT [N][128] f16 (point-major),
// so every neighbor row is one contiguous 256B gather target.
__global__ void prep_transpose(const float* __restrict__ spa, const float* __restrict__ spe,
                               f16* __restrict__ featT, int N) {
  long n = (long)blockIdx.x * blockDim.x + threadIdx.x;
  if (n >= N) return;
  f16* row = featT + n * 128;
  #pragma unroll 8
  for (int c = 0; c < 64; ++c) row[c]      = (f16)spa[(long)c * N + n];
  #pragma unroll 8
  for (int c = 0; c < 64; ++c) row[64 + c] = (f16)spe[(long)c * N + n];
}

// ---------------- main fused kernel ----------------
// Block = 64 threads = 2 independent wave32's; each wave owns one group of
// 16 consecutive points (grid covers N/16 groups exactly).

__global__ __launch_bounds__(64) void fused_cross_att(
    const f16* __restrict__ featT, const f16* __restrict__ w16,
    const long long* __restrict__ nidx,
    const float* __restrict__ spa, const float* __restrict__ spe,
    const float* __restrict__ bVspa, const float* __restrict__ bVspe,
    float* __restrict__ out, int N) {

  __shared__ float qbuf[2][16 * 128];   // Q for 16 points x 128 ch (both attns)
  __shared__ float kv[2][16 * 68];      // K or V tile, padded stride 68 (bank-safe)
  __shared__ float ctxbuf[2][128 * 17]; // fusion output staged for coalesced write
  __shared__ float attnbuf[2][16];

  const int tid    = threadIdx.x;
  const int w      = tid >> 5;
  const int lane   = tid & 31;
  const int lane16 = lane & 15;
  const int hiHalf = lane >> 4;   // 0 | 1
  const int off8   = hiHalf * 8;  // A-frag K offset / C-frag row offset
  const int hi16   = hiHalf * 16; // B-frag K offset

  const int  g  = blockIdx.x * 2 + w;
  const long n0 = (long)g * 16;

  float* Q   = qbuf[w];
  float* KV  = kv[w];
  float* CTX = ctxbuf[w];
  float* ATT = attnbuf[w];

  // ---- Phase 1: Q = feat * WQ^T for all 16 points (WMMA) ----
  const f16* myrowQ = featT + (n0 + lane16) * 128;
  for (int att = 0; att < 2; ++att) {
    const f16* WQ  = w16 + (att * 3 + 0) * 4096;
    const int qoff = att * 64;                       // query channels & Q cols
    v16h a0 = load_afrag(myrowQ, qoff + 0,  off8);
    v16h a1 = load_afrag(myrowQ, qoff + 32, off8);
    for (int nt = 0; nt < 4; ++nt) {
      v8f acc = {};
      v16h b0 = load_bfrag(WQ, nt * 16, lane16, hi16, qoff + 0);
      acc = __builtin_amdgcn_wmma_f32_16x16x32_f16(false, a0, false, b0, (short)0, acc, false, false);
      v16h b1 = load_bfrag(WQ, nt * 16, lane16, hi16, qoff + 32);
      acc = __builtin_amdgcn_wmma_f32_16x16x32_f16(false, a1, false, b1, (short)0, acc, false, false);
      #pragma unroll
      for (int r = 0; r < 8; ++r)   // f32 C/D layout: row r+8*(L/16), col nt*16+L%16
        Q[(r + off8) * 128 + qoff + nt * 16 + lane16] = acc[r];
    }
  }
  __syncthreads();

  // ---- Phase 2: per-point neighbor attention ----
  for (int p = 0; p < 16; ++p) {
    const long n = n0 + p;
    int idxv = 0;
    if (lane < 16) idxv = (int)nidx[n * 16 + lane];
    int idxm = __shfl(idxv, lane16, 32);            // lane L gets index of row m=L%16
    const f16* nbrow = featT + (long)idxm * 128;

    for (int att = 0; att < 2; ++att) {
      const int qoff  = att * 64;        // this attention's query/output channel block
      const int nbase = att ? 0 : 64;    // cross: spa queries attend spe neighbors & v.v.
      const f16* WK   = w16 + (att * 3 + 1) * 4096;
      const f16* WV   = w16 + (att * 3 + 2) * 4096;
      const float* bV = att ? bVspe : bVspa;

      v16h a0 = load_afrag(nbrow, nbase + 0,  off8);  // gathered 16x64 neighbor tile
      v16h a1 = load_afrag(nbrow, nbase + 32, off8);

      // K projection -> KV
      for (int nt = 0; nt < 4; ++nt) {
        v8f acc = {};
        v16h b0 = load_bfrag(WK, nt * 16, lane16, hi16, nbase + 0);
        acc = __builtin_amdgcn_wmma_f32_16x16x32_f16(false, a0, false, b0, (short)0, acc, false, false);
        v16h b1 = load_bfrag(WK, nt * 16, lane16, hi16, nbase + 32);
        acc = __builtin_amdgcn_wmma_f32_16x16x32_f16(false, a1, false, b1, (short)0, acc, false, false);
        #pragma unroll
        for (int r = 0; r < 8; ++r)
          KV[(r + off8) * 68 + nt * 16 + lane16] = acc[r];
      }
      __syncthreads();

      // scores + softmax over 16 neighbors (lane j = L%16; upper half duplicates)
      float s = 0.f;
      const float* qrow = Q + p * 128 + qoff;
      #pragma unroll 8
      for (int d = 0; d < 64; ++d) s += qrow[d] * KV[lane16 * 68 + d];
      s *= 0.125f;                                   // 1/sqrt(64)
      float mx = s;
      #pragma unroll
      for (int m = 1; m <= 8; m <<= 1) mx = fmaxf(mx, __shfl_xor(mx, m, 32));
      float e = __expf(s - mx);
      float sum = e;
      #pragma unroll
      for (int m = 1; m <= 8; m <<= 1) sum += __shfl_xor(sum, m, 32);
      float a_ = e / sum;
      if (lane < 16) ATT[lane] = a_;

      // V projection -> KV (overwrite; in-order DS + barrier below)
      for (int nt = 0; nt < 4; ++nt) {
        v8f acc = {};
        v16h b0 = load_bfrag(WV, nt * 16, lane16, hi16, nbase + 0);
        acc = __builtin_amdgcn_wmma_f32_16x16x32_f16(false, a0, false, b0, (short)0, acc, false, false);
        v16h b1 = load_bfrag(WV, nt * 16, lane16, hi16, nbase + 32);
        acc = __builtin_amdgcn_wmma_f32_16x16x32_f16(false, a1, false, b1, (short)0, acc, false, false);
        #pragma unroll
        for (int r = 0; r < 8; ++r)
          KV[(r + off8) * 68 + nt * 16 + lane16] = acc[r];
      }
      __syncthreads();

      // ctx[d] = bV[d] + sum_j attn[j]*V[j][d]; lane covers d and d+32
      const int d0 = lane, d1 = lane + 32;
      float c0 = bV[d0], c1 = bV[d1];
      #pragma unroll
      for (int j = 0; j < 16; ++j) {
        float aj = ATT[j];
        c0 += aj * KV[j * 68 + d0];
        c1 += aj * KV[j * 68 + d1];
      }
      CTX[(qoff + d0) * 17 + p] = c0;
      CTX[(qoff + d1) * 17 + p] = c1;
      __syncthreads();
    }
  }
  __syncthreads();

  // ---- Phase 3: out = feature + fusion, re-coalesced (16 consecutive n) ----
  for (int i = lane; i < 128 * 16; i += 32) {
    int c = i >> 4, p = i & 15;
    long n = n0 + p;
    float base = (c < 64) ? spa[(long)c * N + n] : spe[(long)(c - 64) * N + n];
    out[(long)c * N + n] = base + CTX[c * 17 + p];
  }
}

// ---------------- launch ----------------

extern "C" void kernel_launch(void* const* d_in, const int* in_sizes, int n_in,
                              void* d_out, int out_size, void* d_ws, size_t ws_size,
                              hipStream_t stream) {
  const float*     spa  = (const float*)d_in[0];
  const float*     spe  = (const float*)d_in[1];
  const long long* nidx = (const long long*)d_in[2];
  const float* WQa = (const float*)d_in[3];
  const float* WKa = (const float*)d_in[4];
  const float* WVa = (const float*)d_in[5];
  const float* bVa = (const float*)d_in[6];
  const float* WQe = (const float*)d_in[7];
  const float* WKe = (const float*)d_in[8];
  const float* WVe = (const float*)d_in[9];
  const float* bVe = (const float*)d_in[10];
  float* out = (float*)d_out;

  const int N = in_sizes[0] / 64;   // 65536

  f16* featT = (f16*)d_ws;                                      // N*128 f16 = 16 MB
  f16* w16   = (f16*)((char*)d_ws + (size_t)N * 128 * sizeof(f16)); // 6*64*64 f16

  prep_weights<<<(6 * 4096 + 255) / 256, 256, 0, stream>>>(WQa, WKa, WVa, WQe, WKe, WVe, w16);
  prep_transpose<<<(N + 255) / 256, 256, 0, stream>>>(spa, spe, featT, N);

  const int groups = N / 16;                 // 4096 groups, 2 waves per block
  fused_cross_att<<<groups / 2, 64, 0, stream>>>(featT, w16, nidx, spa, spe,
                                                 bVa, bVe, out, N);
}
